// GroupViTAssignAttention_15401752723774
// MI455X (gfx1250) — matrix operand order repaired and checked by reading, not compiled, not run on hardware
//
#include <hip/hip_runtime.h>

// ---------------------------------------------------------------------------
// GroupViT assign-attention for MI455X (gfx1250, wave32, WMMA bf16 path)
// B=32, G=64, N=4096, C=384
// Double-buffered LDS pipeline: async global->LDS copy of tile k+1 overlaps
// the WMMA work on tile k (one workgroup barrier per K-step).
// ---------------------------------------------------------------------------

typedef __bf16 bf16;
typedef __attribute__((ext_vector_type(16))) __bf16 v16bf;
typedef __attribute__((ext_vector_type(8)))  float  v8f;

#define B_   32
#define G_   64
#define N_   4096
#define C_   384
#define NT_  (N_ / 128)   // 32 n-tiles in the attention kernel

// ---- gfx1250 async global->LDS path (ASYNCcnt) ----------------------------
#if defined(__has_builtin)
#  if __has_builtin(__builtin_amdgcn_global_load_async_to_lds_b128) && \
      __has_builtin(__builtin_amdgcn_s_wait_asynccnt)
#    define USE_ASYNC_LDS 1
#  endif
#endif
#ifndef USE_ASYNC_LDS
#  define USE_ASYNC_LDS 0
#endif

// Copy 16 bytes global -> LDS (async engine; VDST = per-lane LDS byte addr).
__device__ __forceinline__ void stage_b128(const bf16* __restrict__ gsrc,
                                           bf16* __restrict__ ldst) {
#if USE_ASYNC_LDS
  unsigned lds_addr = (unsigned)(unsigned long long)ldst;  // low 32b = LDS offset
  unsigned long long gaddr = (unsigned long long)gsrc;
  asm volatile("global_load_async_to_lds_b128 %0, %1, off"
               :
               : "v"(lds_addr), "v"(gaddr)
               : "memory");
#else
  *(int4*)ldst = *(const int4*)gsrc;
#endif
}

__device__ __forceinline__ void stage_join() {
#if USE_ASYNC_LDS
  __builtin_amdgcn_s_wait_asynccnt(0);
#endif
}

__device__ __forceinline__ v8f vzero() {
  v8f z;
#pragma unroll
  for (int i = 0; i < 8; ++i) z[i] = 0.0f;
  return z;
}

__device__ __forceinline__ v8f wmma_bf16(v16bf a, v16bf b, v8f c) {
  return __builtin_amdgcn_wmma_f32_16x16x32_bf16(false, a, false, b, (short)0, c,
                                                 false, false);
}

// A fragment (16x32, MxK): lane l -> row = base + (l&15);
// K = (l>>4)*16 .. +15 contiguous (matches B fragment order).
__device__ __forceinline__ v16bf load_frag_a(const bf16* __restrict__ s,
                                             int row_base, int lane, int stride) {
  const bf16* r = s + (size_t)(row_base + (lane & 15)) * stride;
  const int kc = (lane >> 4) * 16;
  v16bf a;
#pragma unroll
  for (int i = 0; i < 16; ++i) a[i] = r[kc + i];
  return a;
}

// B fragment for NT GEMM: lane l -> column n = base + (l&15),
// K = (l>>4)*16 .. +15 contiguous.
__device__ __forceinline__ v16bf load_frag_b_nt(const bf16* __restrict__ s,
                                                int col_base, int lane, int stride) {
  const bf16* r = s + (size_t)(col_base + (lane & 15)) * stride;
  const int kb = (lane >> 4) * 16;
  v16bf b;
#pragma unroll
  for (int i = 0; i < 16; ++i) b[i] = r[kb + i];
  return b;
}

// B fragment for NN GEMM from an LDS tile stored [K][Ncols] (stride = pitch).
__device__ __forceinline__ v16bf load_frag_b_nn(const bf16* __restrict__ s,
                                                int col_base, int lane, int stride) {
  const int c = col_base + (lane & 15);
  const int kb = (lane >> 4) * 16;
  v16bf b;
#pragma unroll
  for (int i = 0; i < 16; ++i) b[i] = s[(size_t)(kb + i) * stride + c];
  return b;
}

// ---------------------------------------------------------------------------
// Generic NT GEMM: Out[m,n] = bias[n] + sum_k A[m,k] * W[n,k]
// A: fp32 or bf16 row-major [M,K]; W: fp32 row-major [N,K] (torch Linear)
// 128x128 tile, BK=32, 8 waves (4 in M x 2 in N), wave = 2x4 16x16 frags.
// ---------------------------------------------------------------------------
template <bool A_F32, bool OUT_BF16>
__global__ __launch_bounds__(256) void gemm_nt_kernel(
    const void* __restrict__ Aptr, const float* __restrict__ Bw,
    const float* __restrict__ bias, void* __restrict__ Out,
    int M, int N, int K) {
  constexpr int BK = 32, LDT = 40;  // padded LDS pitch (80B, 16B-aligned)
  __shared__ __align__(16) bf16 As[2][128 * LDT];  // 2 x 10 KB
  __shared__ __align__(16) bf16 Bs[2][128 * LDT];  // 2 x 10 KB

  const int tid = threadIdx.x;
  const int lane = tid & 31;
  const int wid = tid >> 5;
  const int wave_m = wid & 3;   // 4 waves over 128 rows
  const int wave_n = wid >> 2;  // 2 waves over 128 cols
  const int m0 = blockIdx.x * 128;
  const int n0 = blockIdx.y * 128;

  auto stage_tile = [&](int k0, int buf) {
    if (A_F32) {
      const float* A = (const float*)Aptr;
      if (k0 + 2 * BK < K)  // gfx1250 prefetch hint two tiles ahead
        __builtin_prefetch(A + (size_t)(m0 + (tid >> 1)) * K + k0 + 2 * BK, 0, 1);
#pragma unroll
      for (int it = 0; it < 4; ++it) {
        int s = tid + it * 256;  // 1024 float4 slots = 128 rows x 8 chunks
        int row = s >> 3, kc = (s & 7) * 4;
        float4 f = *(const float4*)(A + (size_t)(m0 + row) * K + k0 + kc);
        bf16* d = &As[buf][row * LDT + kc];
        d[0] = (bf16)f.x; d[1] = (bf16)f.y; d[2] = (bf16)f.z; d[3] = (bf16)f.w;
      }
    } else {
      const bf16* A = (const bf16*)Aptr;
#pragma unroll
      for (int it = 0; it < 2; ++it) {
        int s = tid + it * 256;  // 512 16-byte slots
        int row = s >> 2, kc = (s & 3) * 8;
        stage_b128(A + (size_t)(m0 + row) * K + k0 + kc,
                   &As[buf][row * LDT + kc]);
      }
    }
#pragma unroll
    for (int it = 0; it < 4; ++it) {  // W tile fp32 -> bf16 via VGPRs
      int s = tid + it * 256;
      int row = s >> 3, kc = (s & 7) * 4;
      float4 f = *(const float4*)(Bw + (size_t)(n0 + row) * K + k0 + kc);
      bf16* d = &Bs[buf][row * LDT + kc];
      d[0] = (bf16)f.x; d[1] = (bf16)f.y; d[2] = (bf16)f.z; d[3] = (bf16)f.w;
    }
  };

  v8f acc[2][4];
#pragma unroll
  for (int i = 0; i < 2; ++i)
#pragma unroll
    for (int j = 0; j < 4; ++j) acc[i][j] = vzero();

  const int nk = K / BK;
  stage_tile(0, 0);
  for (int kt = 0; kt < nk; ++kt) {
    const int buf = kt & 1;
    stage_join();     // copies for tile kt complete
    __syncthreads();  // visible to all waves; buf^1 free for re-staging
    if (kt + 1 < nk) stage_tile((kt + 1) * BK, buf ^ 1);  // overlaps WMMA

    v16bf af[2], bfg[4];
#pragma unroll
    for (int i = 0; i < 2; ++i)
      af[i] = load_frag_a(As[buf], wave_m * 32 + i * 16, lane, LDT);
#pragma unroll
    for (int j = 0; j < 4; ++j)
      bfg[j] = load_frag_b_nt(Bs[buf], wave_n * 64 + j * 16, lane, LDT);
#pragma unroll
    for (int i = 0; i < 2; ++i)
#pragma unroll
      for (int j = 0; j < 4; ++j) acc[i][j] = wmma_bf16(af[i], bfg[j], acc[i][j]);
  }

  const int hi8 = (lane >> 4) * 8;  // C/D layout: lanes 16-31 hold M = r+8
#pragma unroll
  for (int i = 0; i < 2; ++i)
#pragma unroll
    for (int j = 0; j < 4; ++j)
#pragma unroll
      for (int r = 0; r < 8; ++r) {
        int m = m0 + wave_m * 32 + i * 16 + r + hi8;
        int n = n0 + wave_n * 64 + j * 16 + (lane & 15);
        float val = acc[i][j][r] + bias[n];
        if (OUT_BF16) ((bf16*)Out)[(size_t)m * N + n] = (bf16)val;
        else          ((float*)Out)[(size_t)m * N + n] = val;
      }
}

// ---------------------------------------------------------------------------
// Fused QK^T + softmax(dim=G) + hard-assign kernel.
// One workgroup = one (batch, 128-column tile of N); full G=64 rows local.
// ---------------------------------------------------------------------------
__global__ __launch_bounds__(256) void attn_softmax_kernel(
    const bf16* __restrict__ Qb, const bf16* __restrict__ Kb,
    float* __restrict__ soft, bf16* __restrict__ T,
    float* __restrict__ rowsumP /* [B, NT, G] */) {
  constexpr int BN = 128, BK = 32, LDT = 40;
  __shared__ __align__(16) bf16 As[2][G_ * LDT];   // 2 x 5 KB
  __shared__ __align__(16) bf16 Bs[2][BN * LDT];   // 2 x 10 KB
  __shared__ float rawS[G_ * BN];                  // 32 KB

  const int tid = threadIdx.x;
  const int lane = tid & 31;
  const int wid = tid >> 5;
  const int wave_m = wid & 1;   // 2 waves over 64 rows
  const int wave_n = wid >> 1;  // 4 waves over 128 cols
  const int b = blockIdx.y;
  const int n0 = blockIdx.x * BN;

  const bf16* Qbase = Qb + (size_t)b * G_ * C_;
  const bf16* Kbase = Kb + ((size_t)b * N_ + n0) * C_;

  auto stage_tile = [&](int k0, int buf) {
    {  // A: 64x32 bf16 = 256 16-byte slots, one per thread
      int row = tid >> 2, kc = (tid & 3) * 8;
      stage_b128(Qbase + (size_t)row * C_ + k0 + kc, &As[buf][row * LDT + kc]);
    }
#pragma unroll
    for (int it = 0; it < 2; ++it) {  // B: 128x32 bf16 = 512 slots
      int s = tid + it * 256;
      int row = s >> 2, kc = (s & 3) * 8;
      stage_b128(Kbase + (size_t)row * C_ + k0 + kc, &Bs[buf][row * LDT + kc]);
    }
  };

  v8f acc[2][2];
#pragma unroll
  for (int i = 0; i < 2; ++i)
#pragma unroll
    for (int j = 0; j < 2; ++j) acc[i][j] = vzero();

  const int nk = C_ / BK;
  stage_tile(0, 0);
  for (int kt = 0; kt < nk; ++kt) {
    const int buf = kt & 1;
    stage_join();
    __syncthreads();
    if (kt + 1 < nk) stage_tile((kt + 1) * BK, buf ^ 1);

    v16bf af[2], bfg[2];
#pragma unroll
    for (int i = 0; i < 2; ++i)
      af[i] = load_frag_a(As[buf], wave_m * 32 + i * 16, lane, LDT);
#pragma unroll
    for (int j = 0; j < 2; ++j)
      bfg[j] = load_frag_b_nt(Bs[buf], wave_n * 32 + j * 16, lane, LDT);
#pragma unroll
    for (int i = 0; i < 2; ++i)
#pragma unroll
      for (int j = 0; j < 2; ++j) acc[i][j] = wmma_bf16(af[i], bfg[j], acc[i][j]);
  }

  const float scale = rsqrtf((float)C_);
  const int hi8 = (lane >> 4) * 8;
#pragma unroll
  for (int i = 0; i < 2; ++i)
#pragma unroll
    for (int j = 0; j < 2; ++j)
#pragma unroll
      for (int r = 0; r < 8; ++r)
        rawS[(wave_m * 32 + i * 16 + r + hi8) * BN +
             wave_n * 32 + j * 16 + (lane & 15)] = acc[i][j][r] * scale;
  __syncthreads();

  if (tid < BN) {  // one thread per column: softmax/argmax over G=64
    const int j = tid;
    float mx = rawS[j];
    int arg = 0;
    for (int g = 1; g < G_; ++g) {
      float x = rawS[g * BN + j];
      if (x > mx) { mx = x; arg = g; }  // first-max tie-break like jnp.argmax
    }
    float se = 0.0f;
    for (int g = 0; g < G_; ++g) se += __expf(rawS[g * BN + j] - mx);
    float inv = 1.0f / se;
    for (int g = 0; g < G_; ++g) {
      float s = __expf(rawS[g * BN + j] - mx) * inv;
      size_t idx = ((size_t)b * G_ + g) * N_ + n0 + j;
      soft[idx] = s;
      float t = ((g == arg ? 1.0f : 0.0f) - s) + s;  // straight-through value
      T[idx] = (bf16)t;
      rawS[g * BN + j] = t;  // reuse tile for deterministic row-sum reduce
    }
  }
  __syncthreads();
  if (tid < G_) {
    float s = 0.0f;
    for (int j = 0; j < BN; ++j) s += rawS[tid * BN + j];
    rowsumP[((size_t)b * NT_ + blockIdx.x) * G_ + tid] = s;
  }
}

// ---------------------------------------------------------------------------
// H = (T @ V) * 1/(rowsum+1): NN GEMM per batch, M=64 (G), N=128 cols of C,
// K=4096. 8 waves (2 in M x 4 in N), each 32x32 = 2x2 frags.
// ---------------------------------------------------------------------------
__global__ __launch_bounds__(256) void tv_gemm_kernel(
    const bf16* __restrict__ T, const bf16* __restrict__ Vb,
    const float* __restrict__ rowsumP, bf16* __restrict__ Hb) {
  constexpr int BN = 128, BK = 32, LDA = 40, LDB = 136;  // 272B pitch, 16B-aligned
  __shared__ __align__(16) bf16 As[2][G_ * LDA];   // 2 x 5 KB
  __shared__ __align__(16) bf16 Bs[2][BK * LDB];   // 2 x 8.5 KB
  __shared__ float invS[G_];

  const int tid = threadIdx.x;
  const int lane = tid & 31;
  const int wid = tid >> 5;
  const int wave_m = wid & 1;
  const int wave_n = wid >> 1;
  const int b = blockIdx.y;
  const int c0 = blockIdx.x * BN;

  if (tid < G_) {  // deterministic reduction of per-tile partial row sums
    float s = 0.0f;
    for (int t = 0; t < NT_; ++t) s += rowsumP[((size_t)b * NT_ + t) * G_ + tid];
    invS[tid] = 1.0f / (s + 1.0f);  // ASSIGN_EPS = 1.0
  }

  const bf16* Tb = T + (size_t)b * G_ * N_;
  const bf16* Vbase = Vb + (size_t)b * N_ * C_;

  auto stage_tile = [&](int k0, int buf) {
    {  // A: T tile 64 x 32
      int row = tid >> 2, kc = (tid & 3) * 8;
      stage_b128(Tb + (size_t)row * N_ + k0 + kc, &As[buf][row * LDA + kc]);
    }
#pragma unroll
    for (int it = 0; it < 2; ++it) {  // B: V tile [32 k-rows][128 c-cols]
      int s = tid + it * 256;
      int kr = s >> 4, cc = (s & 15) * 8;
      stage_b128(Vbase + (size_t)(k0 + kr) * C_ + c0 + cc,
                 &Bs[buf][kr * LDB + cc]);
    }
  };

  v8f acc[2][2];
#pragma unroll
  for (int i = 0; i < 2; ++i)
#pragma unroll
    for (int j = 0; j < 2; ++j) acc[i][j] = vzero();

  const int nk = N_ / BK;
  stage_tile(0, 0);
  for (int kt = 0; kt < nk; ++kt) {
    const int buf = kt & 1;
    stage_join();
    __syncthreads();
    if (kt + 1 < nk) stage_tile((kt + 1) * BK, buf ^ 1);

    v16bf af[2], bfg[2];
#pragma unroll
    for (int i = 0; i < 2; ++i)
      af[i] = load_frag_a(As[buf], wave_m * 32 + i * 16, lane, LDA);
#pragma unroll
    for (int j = 0; j < 2; ++j)
      bfg[j] = load_frag_b_nn(Bs[buf], wave_n * 32 + j * 16, lane, LDB);
#pragma unroll
    for (int i = 0; i < 2; ++i)
#pragma unroll
      for (int j = 0; j < 2; ++j) acc[i][j] = wmma_bf16(af[i], bfg[j], acc[i][j]);
  }

  const int hi8 = (lane >> 4) * 8;
#pragma unroll
  for (int i = 0; i < 2; ++i)
#pragma unroll
    for (int r = 0; r < 8; ++r) {
      int g = wave_m * 32 + i * 16 + r + hi8;
      float inv = invS[g];
#pragma unroll
      for (int j = 0; j < 2; ++j) {
        int c = c0 + wave_n * 32 + j * 16 + (lane & 15);
        Hb[((size_t)b * G_ + g) * C_ + c] = (bf16)(acc[i][j][r] * inv);
      }
    }
}

// ---------------------------------------------------------------------------
extern "C" void kernel_launch(void* const* d_in, const int* in_sizes, int n_in,
                              void* d_out, int out_size, void* d_ws, size_t ws_size,
                              hipStream_t stream) {
  const float* query = (const float*)d_in[0];  // [B,G,C]
  const float* key   = (const float*)d_in[1];  // [B,N,C]
  const float* q_w = (const float*)d_in[2];
  const float* q_b = (const float*)d_in[3];
  const float* k_w = (const float*)d_in[4];
  const float* k_b = (const float*)d_in[5];
  const float* v_w = (const float*)d_in[6];
  const float* v_b = (const float*)d_in[7];
  const float* p_w = (const float*)d_in[8];
  const float* p_b = (const float*)d_in[9];

  // workspace carve-up (all region sizes are 16B multiples)
  char* w = (char*)d_ws;
  bf16* Qb = (bf16*)w; w += (size_t)B_ * G_ * C_ * sizeof(bf16);   // 1.5 MB
  bf16* Kb = (bf16*)w; w += (size_t)B_ * N_ * C_ * sizeof(bf16);   // 100.7 MB
  bf16* Vb = (bf16*)w; w += (size_t)B_ * N_ * C_ * sizeof(bf16);   // 100.7 MB
  bf16* Tt = (bf16*)w; w += (size_t)B_ * G_ * N_ * sizeof(bf16);   // 16.8 MB
  bf16* Hb = (bf16*)w; w += (size_t)B_ * G_ * C_ * sizeof(bf16);   // 1.5 MB
  float* rowsumP = (float*)w;                                      // [B,NT,G]

  float* out  = (float*)d_out;               // [B,G,C] fp32
  float* soft = out + (size_t)B_ * G_ * C_;  // [B,G,N] fp32

  // q/k/v projections (fp32 in -> bf16 out, WMMA bf16)
  gemm_nt_kernel<true, true><<<dim3((B_ * G_) / 128, C_ / 128), 256, 0, stream>>>(
      query, q_w, q_b, Qb, B_ * G_, C_, C_);
  gemm_nt_kernel<true, true><<<dim3((B_ * N_) / 128, C_ / 128), 256, 0, stream>>>(
      key, k_w, k_b, Kb, B_ * N_, C_, C_);
  gemm_nt_kernel<true, true><<<dim3((B_ * N_) / 128, C_ / 128), 256, 0, stream>>>(
      key, v_w, v_b, Vb, B_ * N_, C_, C_);

  // fused QK^T + group-softmax + hard assignment
  attn_softmax_kernel<<<dim3(NT_, B_), 256, 0, stream>>>(Qb, Kb, soft, Tt, rowsumP);

  // H = renormalized attn @ V
  tv_gemm_kernel<<<dim3(C_ / 128, B_), 256, 0, stream>>>(Tt, Vb, rowsumP, Hb);

  // final projection -> fp32 output
  gemm_nt_kernel<false, false><<<dim3((B_ * G_) / 128, C_ / 128), 256, 0, stream>>>(
      Hb, p_w, p_b, out, B_ * G_, C_, C_);
}
